// RuneAligner_44324062495206
// MI455X (gfx1250) — matrix operand order repaired and checked by reading, not compiled, not run on hardware
//
#include <hip/hip_runtime.h>
#include <hip/hip_bf16.h>

typedef __attribute__((ext_vector_type(16))) _Float16 v16h;
typedef __attribute__((ext_vector_type(8)))  float    v8f;

#define Bsz   2048
#define Hdim  128
#define Gdim  512      // 4*H gate columns
#define Lt    48
#define DIN   20
#define D1IN  256      // 2*H layer-1 input

// Fast transcendentals: v_rcp_f32 / v_exp_f32 (TRANS32, co-executes with VALU)
__device__ __forceinline__ float fast_rcp(float x) {
    return __builtin_amdgcn_rcpf(x);
}
__device__ __forceinline__ float fast_sigmoid(float x) {
    return fast_rcp(1.0f + __expf(-x));
}
__device__ __forceinline__ float fast_tanh(float x) {
    // tanh(x) = 1 - 2/(e^{2x}+1); stable for the magnitudes seen here
    float e = __expf(2.0f * x);
    return 1.0f - 2.0f * fast_rcp(e + 1.0f);
}

__device__ __forceinline__ v8f wmma_f16(const v16h& a, const v16h& b, const v8f& c) {
    return __builtin_amdgcn_wmma_f32_16x16x32_f16(false, a, false, b, (short)0, c, false, false);
}

// ---------------------------------------------------------------------------
// Layer 0 LSTM, one direction. Block: 512 threads (16 waves), 16 batch rows.
// Weights held in VGPR fragments across all 48 timesteps; h/c/gates in LDS.
// Writes h (f16) into hcat[b][t][dir*128 + j].
// ---------------------------------------------------------------------------
__global__ __launch_bounds__(512) void lstm0_kernel(
    const float* __restrict__ x,      // (B, DIN, Lt)
    const float* __restrict__ Wih,    // (512, 20)
    const float* __restrict__ Whh,    // (512, 128)
    const float* __restrict__ bias,   // (512)
    _Float16*   __restrict__ hcat,    // (B, Lt, 256)
    int dir)
{
    __shared__ _Float16 xls[16][32];
    __shared__ _Float16 hls[16][Hdim];
    __shared__ float    cls[16][Hdim];
    __shared__ float    gls[16][Gdim];

    const int tid    = threadIdx.x;
    const int wave   = tid >> 5;      // 0..15
    const int lane   = tid & 31;
    const int laneHi = lane >> 4;     // 0/1
    const int lm     = lane & 15;
    const int b0     = blockIdx.x * 16;
    const int n0     = wave * 32;     // 2 N-tiles of 16 cols

    // ---- load persistent B fragments into registers ----
    v16h Bih[2];
    v16h Bhh[2][4];
    float bval[2];
#pragma unroll
    for (int nt = 0; nt < 2; ++nt) {
        const int nglob = n0 + nt * 16 + lm;
        bval[nt] = bias[nglob];
        const float* wih = Wih + (size_t)nglob * DIN;
#pragma unroll
        for (int v = 0; v < 8; ++v) {
            int k = (laneHi ? 16 : 0) + 2 * v;
            Bih[nt][2*v]   = (k     < DIN) ? (_Float16)wih[k]     : (_Float16)0.0f;
            Bih[nt][2*v+1] = (k + 1 < DIN) ? (_Float16)wih[k + 1] : (_Float16)0.0f;
        }
        const float* whh = Whh + (size_t)nglob * Hdim;
#pragma unroll
        for (int kt = 0; kt < 4; ++kt)
#pragma unroll
            for (int v = 0; v < 8; ++v) {
                int k = kt * 32 + (laneHi ? 16 : 0) + 2 * v;
                Bhh[nt][kt][2*v]   = (_Float16)whh[k];
                Bhh[nt][kt][2*v+1] = (_Float16)whh[k + 1];
            }
    }

    // ---- zero h, c ----
    for (int i = tid; i < 16 * Hdim; i += 512) {
        (&hls[0][0])[i] = (_Float16)0.0f;
        (&cls[0][0])[i] = 0.0f;
    }
    __syncthreads();

    for (int t = 0; t < Lt; ++t) {
        const int tt = dir ? (Lt - 1 - t) : t;

        // stage x tile (16 x 20, padded to 32) into LDS as f16
        for (int i = tid; i < 16 * 32; i += 512) {
            int m = i >> 5, k = i & 31;
            float xv = (k < DIN) ? x[(size_t)(b0 + m) * DIN * Lt + (size_t)k * Lt + tt]
                                 : 0.0f;
            xls[m][k] = (_Float16)xv;
        }
        __syncthreads();

        v8f acc[2];
#pragma unroll
        for (int nt = 0; nt < 2; ++nt)
#pragma unroll
            for (int v = 0; v < 8; ++v) acc[nt][v] = bval[nt];

        v16h a;
#pragma unroll
        for (int v = 0; v < 8; ++v) {
            int k0 = laneHi * 16 + 2 * v;
            a[2*v]   = xls[lm][k0];
            a[2*v+1] = xls[lm][k0 + 1];
        }
        acc[0] = wmma_f16(a, Bih[0], acc[0]);
        acc[1] = wmma_f16(a, Bih[1], acc[1]);

#pragma unroll
        for (int kt = 0; kt < 4; ++kt) {
#pragma unroll
            for (int v = 0; v < 8; ++v) {
                int k0 = kt * 32 + laneHi * 16 + 2 * v;
                a[2*v]   = hls[lm][k0];
                a[2*v+1] = hls[lm][k0 + 1];
            }
            acc[0] = wmma_f16(a, Bhh[0][kt], acc[0]);
            acc[1] = wmma_f16(a, Bhh[1][kt], acc[1]);
        }

#pragma unroll
        for (int nt = 0; nt < 2; ++nt)
#pragma unroll
            for (int v = 0; v < 8; ++v) {
                int m = v + laneHi * 8;
                gls[m][n0 + nt * 16 + lm] = acc[nt][v];
            }
        __syncthreads();

        // elementwise LSTM cell update; 2048 elements / 512 threads
        for (int e = tid; e < 16 * Hdim; e += 512) {
            int m = e >> 7, j = e & 127;
            float ig = gls[m][j];
            float fg = gls[m][j + 128];
            float gg = gls[m][j + 256];
            float og = gls[m][j + 384];
            float c  = cls[m][j];
            c = fast_sigmoid(fg) * c + fast_sigmoid(ig) * fast_tanh(gg);
            float h = fast_sigmoid(og) * fast_tanh(c);
            cls[m][j] = c;
            hls[m][j] = (_Float16)h;
            hcat[((size_t)(b0 + m) * Lt + tt) * D1IN + dir * Hdim + j] = (_Float16)h;
        }
        __syncthreads();
    }
}

// ---------------------------------------------------------------------------
// Layer 1 LSTM, one direction, fused with the sim_W projection:
// accumulates u[b][t] += 0.5 * sum_j h[b][t][j] * simW[dir*128 + j]
// ---------------------------------------------------------------------------
__global__ __launch_bounds__(512) void lstm1_kernel(
    const _Float16* __restrict__ hin,  // (B, Lt, 256)
    const float* __restrict__ Wih,     // (512, 256)
    const float* __restrict__ Whh,     // (512, 128)
    const float* __restrict__ bias,    // (512)
    const float* __restrict__ simW,    // (256)
    float*       __restrict__ uout,    // (B, Lt)
    int dir, int addmode)
{
    __shared__ _Float16 xls[16][D1IN];
    __shared__ _Float16 hls[16][Hdim];
    __shared__ float    cls[16][Hdim];
    __shared__ float    gls[16][Gdim];

    const int tid    = threadIdx.x;
    const int wave   = tid >> 5;
    const int lane   = tid & 31;
    const int laneHi = lane >> 4;
    const int lm     = lane & 15;
    const int b0     = blockIdx.x * 16;
    const int n0     = wave * 32;

    v16h Bih[2][8];
    v16h Bhh[2][4];
    float bval[2];
#pragma unroll
    for (int nt = 0; nt < 2; ++nt) {
        const int nglob = n0 + nt * 16 + lm;
        bval[nt] = bias[nglob];
        const float* wih = Wih + (size_t)nglob * D1IN;
#pragma unroll
        for (int kt = 0; kt < 8; ++kt)
#pragma unroll
            for (int v = 0; v < 8; ++v) {
                int k = kt * 32 + (laneHi ? 16 : 0) + 2 * v;
                Bih[nt][kt][2*v]   = (_Float16)wih[k];
                Bih[nt][kt][2*v+1] = (_Float16)wih[k + 1];
            }
        const float* whh = Whh + (size_t)nglob * Hdim;
#pragma unroll
        for (int kt = 0; kt < 4; ++kt)
#pragma unroll
            for (int v = 0; v < 8; ++v) {
                int k = kt * 32 + (laneHi ? 16 : 0) + 2 * v;
                Bhh[nt][kt][2*v]   = (_Float16)whh[k];
                Bhh[nt][kt][2*v+1] = (_Float16)whh[k + 1];
            }
    }

    // per-lane simW values for the fused reduction (row = wave)
    float sw[4];
#pragma unroll
    for (int q = 0; q < 4; ++q) sw[q] = simW[dir * Hdim + lane + q * 32];

    for (int i = tid; i < 16 * Hdim; i += 512) {
        (&hls[0][0])[i] = (_Float16)0.0f;
        (&cls[0][0])[i] = 0.0f;
    }
    __syncthreads();

    for (int t = 0; t < Lt; ++t) {
        const int tt = dir ? (Lt - 1 - t) : t;

        // stage input tile: 16 rows x 256 f16, 16B vector loads
        {
            // 16*256*2 bytes = 8192 B = 512 x uint4
            int m     = tid >> 5;         // 16 rows, 32 chunks each
            int chunk = tid & 31;
            const uint4* src = (const uint4*)(hin + ((size_t)(b0 + m) * Lt + tt) * D1IN);
            ((uint4*)&xls[m][0])[chunk] = src[chunk];
        }
        __syncthreads();

        v8f acc[2];
#pragma unroll
        for (int nt = 0; nt < 2; ++nt)
#pragma unroll
            for (int v = 0; v < 8; ++v) acc[nt][v] = bval[nt];

        v16h a;
#pragma unroll
        for (int kt = 0; kt < 8; ++kt) {
#pragma unroll
            for (int v = 0; v < 8; ++v) {
                int k0 = kt * 32 + laneHi * 16 + 2 * v;
                a[2*v]   = xls[lm][k0];
                a[2*v+1] = xls[lm][k0 + 1];
            }
            acc[0] = wmma_f16(a, Bih[0][kt], acc[0]);
            acc[1] = wmma_f16(a, Bih[1][kt], acc[1]);
        }
#pragma unroll
        for (int kt = 0; kt < 4; ++kt) {
#pragma unroll
            for (int v = 0; v < 8; ++v) {
                int k0 = kt * 32 + laneHi * 16 + 2 * v;
                a[2*v]   = hls[lm][k0];
                a[2*v+1] = hls[lm][k0 + 1];
            }
            acc[0] = wmma_f16(a, Bhh[0][kt], acc[0]);
            acc[1] = wmma_f16(a, Bhh[1][kt], acc[1]);
        }

#pragma unroll
        for (int nt = 0; nt < 2; ++nt)
#pragma unroll
            for (int v = 0; v < 8; ++v) {
                int m = v + laneHi * 8;
                gls[m][n0 + nt * 16 + lm] = acc[nt][v];
            }
        __syncthreads();

        for (int e = tid; e < 16 * Hdim; e += 512) {
            int m = e >> 7, j = e & 127;
            float ig = gls[m][j];
            float fg = gls[m][j + 128];
            float gg = gls[m][j + 256];
            float og = gls[m][j + 384];
            float c  = cls[m][j];
            c = fast_sigmoid(fg) * c + fast_sigmoid(ig) * fast_tanh(gg);
            float h = fast_sigmoid(og) * fast_tanh(c);
            cls[m][j] = c;
            hls[m][j] = (_Float16)h;
        }
        __syncthreads();

        // fused 0.5 * h . simW  (wave w reduces row w)
        {
            float p = 0.0f;
#pragma unroll
            for (int q = 0; q < 4; ++q)
                p += (float)hls[wave][lane + q * 32] * sw[q];
#pragma unroll
            for (int off = 16; off > 0; off >>= 1)
                p += __shfl_xor(p, off, 32);
            if (lane == 0) {
                float val = 0.5f * p;
                size_t idx = (size_t)(b0 + wave) * Lt + tt;
                if (addmode) uout[idx] += val;
                else         uout[idx]  = val;
            }
        }
        __syncthreads();
    }
}

// ---------------------------------------------------------------------------
// Needleman-Wunsch scoring (cummax formulation), one thread per batch element.
// sim(i,j) = leaky_relu(u[i-1] + v[j-1] + sim_b, 0.01)
// ---------------------------------------------------------------------------
__global__ __launch_bounds__(128) void nw_kernel(
    const float* __restrict__ u,
    const float* __restrict__ v,
    const float* __restrict__ simb_p,
    const float* __restrict__ gap_p,
    const int*   __restrict__ len_neg,
    float*       __restrict__ out,
    int Bn)
{
    __shared__ float prev[Lt + 1][128];
    const int tid = threadIdx.x;
    const int b   = blockIdx.x * 128 + tid;
    if (b >= Bn) return;

    const float gap = *gap_p;
    const float sb  = *simb_p;
    const float* ub = u + (size_t)b * Lt;
    const float* vb = v + (size_t)b * Lt;

    for (int j = 0; j <= Lt; ++j) prev[j][tid] = gap * (float)j;

    float last = gap * (float)Lt;
    for (int i = 1; i <= Lt; ++i) {
        float ui      = ub[i - 1];
        float pending = gap * (float)i;     // cur_0
        float run     = pending;            // cummax state (seq - gap*j)
        float pjm1    = prev[0][tid];
        for (int j = 1; j <= Lt; ++j) {
            float pj = prev[j][tid];
            float s  = ui + vb[j - 1] + sb;
            s = (s > 0.0f) ? s : 0.01f * s;                 // leaky_relu
            float cand = fmaxf(pjm1 + s, pj + gap);
            run = fmaxf(run, cand - gap * (float)j);
            float curj = run + gap * (float)j;
            prev[j - 1][tid] = pending;
            pending = curj;
            pjm1 = pj;
        }
        prev[Lt][tid] = pending;
        last = pending;
    }

    int ln = len_neg[b];
    if (ln < 1) ln = 1;
    out[b] = last / (float)ln;
}

// ---------------------------------------------------------------------------
extern "C" void kernel_launch(void* const* d_in, const int* in_sizes, int n_in,
                              void* d_out, int out_size, void* d_ws, size_t ws_size,
                              hipStream_t stream)
{
    const float* bneg = (const float*)d_in[0];
    const int*   lneg = (const int*)d_in[1];
    const float* bpos = (const float*)d_in[2];
    // d_in[3] = len_pos (unused by reference output)
    const float* Wih0 = (const float*)d_in[4];   // (2, 512, 20)
    const float* Whh0 = (const float*)d_in[5];   // (2, 512, 128)
    const float* b0p  = (const float*)d_in[6];   // (2, 512)
    const float* Wih1 = (const float*)d_in[7];   // (2, 512, 256)
    const float* Whh1 = (const float*)d_in[8];   // (2, 512, 128)
    const float* b1p  = (const float*)d_in[9];   // (2, 512)
    const float* simW = (const float*)d_in[10];  // (256)
    const float* simb = (const float*)d_in[11];  // (1)
    const float* gap  = (const float*)d_in[12];  // (1)

    const int Bn = in_sizes[1];                  // 2048

    // workspace layout
    _Float16* hcat = (_Float16*)d_ws;                                  // B*Lt*256 f16
    float*    ubuf = (float*)((char*)d_ws + (size_t)Bn * Lt * D1IN * sizeof(_Float16));
    float*    vbuf = ubuf + (size_t)Bn * Lt;

    const dim3 lgrid(Bn / 16);
    const dim3 lblk(512);

    const float* xin[2]  = { bneg, bpos };
    float*       uv[2]   = { ubuf, vbuf };

    for (int s = 0; s < 2; ++s) {
        // layer 0, both directions (disjoint halves of hcat)
        lstm0_kernel<<<lgrid, lblk, 0, stream>>>(
            xin[s], Wih0, Whh0, b0p, hcat, 0);
        lstm0_kernel<<<lgrid, lblk, 0, stream>>>(
            xin[s], Wih0 + (size_t)Gdim * DIN, Whh0 + (size_t)Gdim * Hdim,
            b0p + Gdim, hcat, 1);
        // layer 1, fused sim_W projection
        lstm1_kernel<<<lgrid, lblk, 0, stream>>>(
            hcat, Wih1, Whh1, b1p, simW, uv[s], 0, 0);
        lstm1_kernel<<<lgrid, lblk, 0, stream>>>(
            hcat, Wih1 + (size_t)Gdim * D1IN, Whh1 + (size_t)Gdim * Hdim,
            b1p + Gdim, simW, uv[s], 1, 1);
    }

    nw_kernel<<<dim3((Bn + 127) / 128), dim3(128), 0, stream>>>(
        ubuf, vbuf, simb, gap, lneg, (float*)d_out, Bn);
}
